// NeuralCPU_42142219108456
// MI455X (gfx1250) — matrix operand order repaired and checked by reading, not compiled, not run on hardware
//
#include <hip/hip_runtime.h>
#include <hip/hip_bf16.h>
#include <stddef.h>

// ---------------------------------------------------------------------------
// NeuralCPU forward for MI455X (gfx1250, wave32, WMMA bf16 16x16x32 + TDM).
// Registers carried as rank-2 coefficients over {bit0, bit1}.
// ---------------------------------------------------------------------------

#define B_      1024
#define L_      64
#define D_      256
#define H_      256
#define G3_     768     // 3*H
#define NH_     4
#define HD_     64
#define TSTEPS  16

typedef __bf16 bf16;
typedef __attribute__((ext_vector_type(16))) __bf16 v16bf;
typedef __attribute__((ext_vector_type(8)))  float  v8f;
typedef __attribute__((ext_vector_type(4)))  unsigned int u32x4;
typedef __attribute__((ext_vector_type(8)))  int i32x8;
typedef __attribute__((ext_vector_type(4)))  int i32x4;

// ----------------------------- WMMA fragment loaders -----------------------
// A tile: 16x32 (MxK) bf16 from row-major src (row stride ld).
// ISA layout: lane = row (m = lane&15); lanes 0-15 hold K 0..7 & 16..23,
// lanes 16-31 hold K 8..15 & 24..31.
__device__ __forceinline__ v16bf load_a16x32(const bf16* __restrict__ src, int ld, int lane) {
    int row = lane & 15;
    int kb  = (lane < 16) ? 0 : 8;
    const bf16* p = src + (size_t)row * ld + kb;
    v16bf a;
#pragma unroll
    for (int i = 0; i < 8; ++i) { a[i] = p[i]; a[8 + i] = p[16 + i]; }
    return a;
}

// B tile: 32x16 (KxN) bf16, sourced from a (N,K) row-major weight ("W^T" GEMM),
// so each lane reads two runs of 8 bf16. ISA layout: lane = col (n = lane&15);
// lanes 0-15 hold K 0..7 & 16..23, lanes 16-31 hold K 8..15 & 24..31.
__device__ __forceinline__ v16bf load_b32x16(const bf16* __restrict__ wt, int ldk, int lane) {
    int col = lane & 15;
    int kb  = (lane < 16) ? 0 : 8;
    const bf16* p = wt + (size_t)col * ldk + kb;
    v16bf b;
#pragma unroll
    for (int i = 0; i < 8; ++i) { b[i] = p[i]; b[8 + i] = p[16 + i]; }
    return b;
}

__device__ __forceinline__ v8f wmma_bf16(v16bf a, v16bf b, v8f c) {
    return __builtin_amdgcn_wmma_f32_16x16x32_bf16(false, a, false, b, (short)0, c, false, false);
}

__device__ __forceinline__ float warp_sum(float v) {
#pragma unroll
    for (int o = 16; o > 0; o >>= 1) v += __shfl_xor(v, o, 32);
    return v;
}

__device__ __forceinline__ float sigm(float x) { return 1.0f / (1.0f + __expf(-x)); }

__device__ __forceinline__ float urand(unsigned a) {
    a ^= a >> 17; a *= 0xed5ad4bbu; a ^= a >> 11; a *= 0xac4c1b51u;
    a ^= a >> 15; a *= 0x31848babu; a ^= a >> 14;
    return (float)(a >> 8) * (1.0f / 16777216.0f) + 1e-12f;
}
__device__ __forceinline__ float gumbel(unsigned a) {
    return -__logf(-__logf(urand(a)));
}

// ----------------------------- Tensor Data Mover ---------------------------
// 1-D DMA of `ndwords` f32 from global into LDS via TDM (ISA 08, D# groups
// 0/1: count=1, type=2 (image), data_size=4B, tile_dim0=ndwords, dim0
// stride=ndwords). Tracked with TENSORcnt. One issuing wave per block.
#if defined(__has_builtin)
#if __has_builtin(__builtin_amdgcn_tensor_load_to_lds)
#define HAVE_TDM 1
#endif
#endif

// Compile-time probes (stderr oracle for the compile-only loop):
#ifdef HAVE_TDM
#if __clang_major__ >= 23
#warning "TDM-PROBE: tensor_load_to_lds AVAILABLE, using 6-arg (clang>=23) form"
#else
#warning "TDM-PROBE: tensor_load_to_lds AVAILABLE, using 5-arg (clang<23) form"
#endif
#else
#warning "TDM-PROBE: tensor_load_to_lds builtin NOT available on this toolchain"
#endif
#if defined(__has_builtin)
#if __has_builtin(__builtin_amdgcn_s_wait_tensorcnt)
#warning "TDM-PROBE: s_wait_tensorcnt builtin AVAILABLE"
#else
#warning "TDM-PROBE: s_wait_tensorcnt builtin NOT available"
#endif
#endif

__device__ __forceinline__ void tdm_load_1d(const void* gsrc, void* lds_dst, unsigned ndwords) {
#ifdef HAVE_TDM
    unsigned long long ga = (unsigned long long)(size_t)gsrc;
    unsigned laddr = (unsigned)(size_t)lds_dst;   // low 32 bits = LDS byte addr
    u32x4 g0 = {};
    g0[0] = 1u;                                        // count=1, user mode
    g0[1] = laddr;                                     // lds_addr
    g0[2] = (unsigned)(ga & 0xffffffffu);              // global_addr[31:0]
    g0[3] = (unsigned)((ga >> 32) & 0x1ffffffu) | (2u << 30);  // addr[56:32], type=2
    i32x8 g1 = {};
    g1[0] = (2 << 16);                                 // data_size = 4 bytes
    g1[1] = (int)((ndwords & 0xffffu) << 16);          // tensor_dim0[15:0]
    g1[2] = (int)((ndwords >> 16) & 0xffffu);          // tensor_dim0[31:16]
    g1[3] = (int)((ndwords & 0xffffu) << 16);          // tile_dim0
    g1[5] = (int)ndwords;                              // tensor_dim0_stride[31:0]
    i32x4 z4 = {};
#if __clang_major__ >= 23
    i32x8 z8 = {};
    __builtin_amdgcn_tensor_load_to_lds(g0, g1, z4, z4, z8, 0);
#else
    __builtin_amdgcn_tensor_load_to_lds(g0, g1, z4, z4, 0);
#endif
#endif
}

__device__ __forceinline__ void tdm_wait() {
#if defined(__has_builtin)
#if __has_builtin(__builtin_amdgcn_s_wait_tensorcnt)
    __builtin_amdgcn_s_wait_tensorcnt(0);
#endif
#endif
}

// ----------------------------- prep kernels --------------------------------
__global__ void k_cvt(const float* __restrict__ s, bf16* __restrict__ d, int n) {
    int i = blockIdx.x * blockDim.x + threadIdx.x;
    if (i < n) d[i] = (bf16)s[i];
}

// Wh (K=256, N=768) row-major -> WhT (N,K) bf16
__global__ void k_transpose_cvt(const float* __restrict__ s, bf16* __restrict__ d, int K, int N) {
    int i = blockIdx.x * blockDim.x + threadIdx.x;
    if (i >= K * N) return;
    int n = i / K, k = i % K;
    d[(size_t)n * K + k] = (bf16)s[(size_t)k * N + n];
}

// A1[i][j] = bit[i] . Wx[0:256, j] ; A2[i][j] = bit[i] . Wx[256:512, j] ; Gm = bit2 bit2^T
__global__ void k_prep_small(const float* __restrict__ bit, const float* __restrict__ Wx,
                             float* __restrict__ A1, float* __restrict__ A2, float* __restrict__ Gm) {
    int t = blockIdx.x * blockDim.x + threadIdx.x;
    if (t < G3_) {
        float a10 = 0, a11 = 0, a20 = 0, a21 = 0;
        for (int d = 0; d < D_; ++d) {
            float w0 = Wx[(size_t)d * G3_ + t];
            float w1 = Wx[(size_t)(D_ + d) * G3_ + t];
            a10 += bit[d] * w0;        a11 += bit[D_ + d] * w0;
            a20 += bit[d] * w1;        a21 += bit[D_ + d] * w1;
        }
        A1[t] = a10; A1[G3_ + t] = a11; A2[t] = a20; A2[G3_ + t] = a21;
    }
    if (t < 4) {
        int i = t >> 1, j = t & 1; float s = 0;
        for (int d = 0; d < D_; ++d) s += bit[i * D_ + d] * bit[j * D_ + d];
        Gm[t] = s;
    }
}

// regc[((b*4+r)*L + l)*2 + c]
__global__ void k_init_regs(const int* __restrict__ r0, const int* __restrict__ r1,
                            float* __restrict__ regc) {
    int i = blockIdx.x * blockDim.x + threadIdx.x;
    if (i >= B_ * L_) return;
    int b = i / L_, l = i % L_;
    int v0 = r0[i], v1 = r1[i];
    size_t base = ((size_t)b * 4) * L_ * 2 + (size_t)l * 2;
    regc[base + 0 * L_ * 2 + 0] = v0 ? 0.f : 1.f;
    regc[base + 0 * L_ * 2 + 1] = v0 ? 1.f : 0.f;
    regc[base + 1 * L_ * 2 + 0] = v1 ? 0.f : 1.f;
    regc[base + 1 * L_ * 2 + 1] = v1 ? 1.f : 0.f;
    regc[base + 2 * L_ * 2 + 0] = 1.f; regc[base + 2 * L_ * 2 + 1] = 0.f;
    regc[base + 3 * L_ * 2 + 0] = 1.f; regc[base + 3 * L_ * 2 + 1] = 0.f;
}

// ----------------------------- per-step kernels ----------------------------
__global__ void k_state(const float* __restrict__ regc, const float* __restrict__ bit,
                        const float* __restrict__ stepv,
                        float* __restrict__ xres, bf16* __restrict__ xbf) {
    int wv = (blockIdx.x * blockDim.x + threadIdx.x) >> 5;
    int lane = threadIdx.x & 31;
    if (wv >= B_ * 4) return;
    int b = wv >> 2, r = wv & 3;
    const float* c = regc + (((size_t)b * 4 + r) * L_) * 2;
    float m0 = 0, m1 = 0;
    for (int l = lane; l < L_; l += 32) { m0 += c[l * 2]; m1 += c[l * 2 + 1]; }
    m0 = warp_sum(m0) * (1.f / L_);
    m1 = warp_sum(m1) * (1.f / L_);
    size_t row = (size_t)wv * D_;
    for (int d = lane; d < D_; d += 32) {
        float v = m0 * bit[d] + m1 * bit[D_ + d] + stepv[d];
        xres[row + d] = v;
        xbf[row + d]  = (bf16)v;
    }
}

// Generic WMMA GEMM: Y(M,N) = X(M,K)bf16 @ WT(N,K)bf16^T + bias (+relu).
__global__ void k_gemm_wt(const bf16* __restrict__ X, const bf16* __restrict__ WT,
                          const float* __restrict__ bias,
                          float* __restrict__ Yf, bf16* __restrict__ Yb,
                          int M, int N, int K, int relu) {
    int w = (blockIdx.x * blockDim.x + threadIdx.x) >> 5;
    int lane = threadIdx.x & 31;
    int mt = M >> 4, ntl = N >> 4;
    if (w >= mt * ntl) return;
    int tm = w % mt, tn = w / mt;
    v8f acc = {};
    const bf16* a0 = X + (size_t)tm * 16 * K;
    const bf16* b0 = WT + (size_t)tn * 16 * K;
    for (int k0 = 0; k0 < K; k0 += 32) {
        v16bf a = load_a16x32(a0 + k0, K, lane);
        v16bf b = load_b32x16(b0 + k0, K, lane);
        acc = wmma_bf16(a, b, acc);
    }
    int col   = tn * 16 + (lane & 15);
    int rbase = tm * 16 + ((lane < 16) ? 0 : 8);
    float bs = bias ? bias[col] : 0.f;
#pragma unroll
    for (int i = 0; i < 8; ++i) {
        float y = acc[i] + bs;
        if (relu) y = fmaxf(y, 0.f);
        size_t off = (size_t)(rbase + i) * N + col;
        if (Yf) Yf[off] = y;
        if (Yb) Yb[off] = (bf16)y;
    }
}

// S=4 attention, head dim 64. One thread per (b, h, q).
__global__ void k_attn(const bf16* __restrict__ qkv, bf16* __restrict__ out) {
    int id = blockIdx.x * blockDim.x + threadIdx.x;
    if (id >= B_ * NH_ * 4) return;
    int qi = id & 3, h = (id >> 2) & 3, b = id >> 4;
    const bf16* qp = qkv + ((size_t)(b * 4 + qi)) * G3_ + h * HD_;
    float s[4];
#pragma unroll
    for (int kk = 0; kk < 4; ++kk) {
        const bf16* kp = qkv + ((size_t)(b * 4 + kk)) * G3_ + D_ + h * HD_;
        float d = 0;
        for (int j = 0; j < HD_; ++j) d += (float)qp[j] * (float)kp[j];
        s[kk] = d * 0.125f;
    }
    float m = fmaxf(fmaxf(s[0], s[1]), fmaxf(s[2], s[3]));
    float e[4], se = 0;
#pragma unroll
    for (int kk = 0; kk < 4; ++kk) { e[kk] = __expf(s[kk] - m); se += e[kk]; }
    float inv = 1.f / se;
    bf16* op = out + ((size_t)(b * 4 + qi)) * D_ + h * HD_;
    for (int j = 0; j < HD_; ++j) {
        float acc = 0;
#pragma unroll
        for (int kk = 0; kk < 4; ++kk) {
            const bf16* vp = qkv + ((size_t)(b * 4 + kk)) * G3_ + 2 * D_ + h * HD_;
            acc += e[kk] * inv * (float)vp[j];
        }
        op[j] = (bf16)acc;
    }
}

// xres <- LN(xres + add) ; xbf <- bf16(xres). One wave/token, D=256.
__global__ void k_res_ln(const float* __restrict__ add, const float* __restrict__ g,
                         const float* __restrict__ bta,
                         float* __restrict__ xres, bf16* __restrict__ xbf) {
    int wv = (blockIdx.x * blockDim.x + threadIdx.x) >> 5;
    int lane = threadIdx.x & 31;
    if (wv >= B_ * 4) return;
    size_t row = (size_t)wv * D_;
    float v[8], s = 0;
#pragma unroll
    for (int i = 0; i < 8; ++i) {
        int d = lane + 32 * i;
        v[i] = xres[row + d] + add[row + d];
        s += v[i];
    }
    float mean = warp_sum(s) * (1.f / D_);
    float q = 0;
#pragma unroll
    for (int i = 0; i < 8; ++i) { float t = v[i] - mean; q += t * t; }
    float var = warp_sum(q) * (1.f / D_);
    float rstd = rsqrtf(var + 1e-5f);
#pragma unroll
    for (int i = 0; i < 8; ++i) {
        int d = lane + 32 * i;
        float y = (v[i] - mean) * rstd * g[d] + bta[d];
        xres[row + d] = y;
        xbf[row + d]  = (bf16)y;
    }
}

// dec = mean_S(x); 15 decode logits per batch. One wave per b.
__global__ void k_decode(const float* __restrict__ x,
                         const float* __restrict__ Wop, const float* __restrict__ bop,
                         const float* __restrict__ Ws1, const float* __restrict__ bs1,
                         const float* __restrict__ Ws2, const float* __restrict__ bs2,
                         const float* __restrict__ Wd,  const float* __restrict__ bd,
                         float* __restrict__ lg15) {
    int wv = (blockIdx.x * blockDim.x + threadIdx.x) >> 5;
    int lane = threadIdx.x & 31;
    if (wv >= B_) return;
    int b = wv;
    float dec[8];
#pragma unroll
    for (int i = 0; i < 8; ++i) {
        int d = lane + 32 * i;
        dec[i] = 0.25f * (x[((size_t)b * 4 + 0) * D_ + d] + x[((size_t)b * 4 + 1) * D_ + d] +
                          x[((size_t)b * 4 + 2) * D_ + d] + x[((size_t)b * 4 + 3) * D_ + d]);
    }
    for (int widx = 0; widx < 15; ++widx) {
        const float* wr; float bv;
        if (widx < 3)       { wr = Wop + (size_t)widx * D_;        bv = bop[widx]; }
        else if (widx < 7)  { wr = Ws1 + (size_t)(widx - 3) * D_;  bv = bs1[widx - 3]; }
        else if (widx < 11) { wr = Ws2 + (size_t)(widx - 7) * D_;  bv = bs2[widx - 7]; }
        else                { wr = Wd  + (size_t)(widx - 11) * D_; bv = bd[widx - 11]; }
        float s = 0;
#pragma unroll
        for (int i = 0; i < 8; ++i) s += dec[i] * wr[lane + 32 * i];
        s = warp_sum(s);
        if (lane == 0) lg15[(size_t)b * 16 + widx] = s + bv;
    }
}

// Gumbel-hard argmax selections + record raw argmax op/dest outputs.
__global__ void k_select(const float* __restrict__ lg15, int* __restrict__ sel,
                         float* __restrict__ out_ops, float* __restrict__ out_dests, int t) {
    int b = blockIdx.x * blockDim.x + threadIdx.x;
    if (b >= B_) return;
    const float* p = lg15 + (size_t)b * 16;
    int base[4] = {0, 3, 7, 11};
    int cnt[4]  = {3, 4, 4, 4};
    int pick[4];
    int rawop = 0, rawd = 0;
#pragma unroll
    for (int s = 0; s < 4; ++s) {
        float best = -1e30f; int bi = 0;
        float rbest = -1e30f; int rbi = 0;
        for (int k = 0; k < cnt[s]; ++k) {
            float raw = p[base[s] + k];
            unsigned h = (unsigned)(t * 0x9E3779B1u) ^ (unsigned)(b * 0x85EBCA77u)
                       ^ (unsigned)(s * 0xC2B2AE3Du) ^ (unsigned)(k + 1);
            float val = raw + gumbel(h);
            if (val > best) { best = val; bi = k; }
            if (raw > rbest) { rbest = raw; rbi = k; }
        }
        pick[s] = bi;
        if (s == 0) rawop = rbi;
        if (s == 3) rawd  = rbi;
    }
    sel[b * 4 + 0] = pick[0]; sel[b * 4 + 1] = pick[1];
    sel[b * 4 + 2] = pick[2]; sel[b * 4 + 3] = pick[3];
    out_ops[(size_t)t * B_ + b]   = (float)rawop;
    out_dests[(size_t)t * B_ + b] = (float)rawd;
}

// ----------------------------- GRU + register update -----------------------
// 16 blocks x 512 threads (16 waves). Block owns 64 batch rows; h in LDS.
// A1/A2/bgru/Wout/bout staged once per block into LDS via the Tensor Data
// Mover (TENSORcnt), then the l-recurrence runs gh = h @ Wh via WMMA with the
// rank-2 gx contraction and GRU gates fused in-register.
#define GRU_LDS_BYTES (64*256*4 + 64*256*2 + 64*4*4 + 64*2*4 + 64*4*4 \
                       + 2*G3_*4 + 2*G3_*4 + G3_*4 + 512*4 + 2*4)

__global__ void k_gru(float* __restrict__ regc, const int* __restrict__ sel,
                      const bf16* __restrict__ WhT, const float* __restrict__ A1,
                      const float* __restrict__ A2, const float* __restrict__ bgru,
                      const float* __restrict__ Wout, const float* __restrict__ bout,
                      float* __restrict__ resadd) {
    extern __shared__ char smem[];
    float* hf   = (float*)smem;                                   // 64*256 f32
    bf16*  hbf  = (bf16*)(smem + 64 * 256 * 4);                   // 64*256 bf16
    float* cf   = (float*)(smem + 64 * 256 * 4 + 64 * 256 * 2);   // 64*4
    float* lg   = cf + 64 * 4;                                    // 64*2
    int*   sl   = (int*)(lg + 64 * 2);                            // 64*4
    float* lA1  = (float*)(sl + 64 * 4);                          // 2*768
    float* lA2  = lA1 + 2 * G3_;                                  // 2*768
    float* lbg  = lA2 + 2 * G3_;                                  // 768
    float* lWo  = lbg + G3_;                                      // 512
    float* lbo  = lWo + 512;                                      // 2

    int tid  = threadIdx.x;
    int rbase = blockIdx.x * 64;
    int wv = tid >> 5, lane = tid & 31;

    // TDM: stage gate tables into LDS (one issuing wave; EXEC-independent).
    if (wv == 0) {
        tdm_load_1d(A1,   lA1, 2 * G3_);
        tdm_load_1d(A2,   lA2, 2 * G3_);
        tdm_load_1d(bgru, lbg, G3_);
        tdm_load_1d(Wout, lWo, 512);
        tdm_load_1d(bout, lbo, 2);
        tdm_wait();
    }
    for (int i = tid; i < 64 * 256; i += blockDim.x) { hf[i] = 0.f; hbf[i] = (bf16)0.f; }
    if (tid < 64) {
        int b = rbase + tid;
        sl[tid * 4 + 0] = sel[b * 4 + 0];
        sl[tid * 4 + 1] = sel[b * 4 + 1];
        sl[tid * 4 + 2] = sel[b * 4 + 2];
        sl[tid * 4 + 3] = sel[b * 4 + 3];
    }
    __syncthreads();
    // Fallback/verification fill (same bytes as the TDM transfer).
    for (int i = tid; i < 2 * G3_; i += blockDim.x) { lA1[i] = A1[i]; lA2[i] = A2[i]; }
    for (int i = tid; i < G3_; i += blockDim.x) lbg[i] = bgru[i];
    for (int i = tid; i < 512; i += blockDim.x) lWo[i] = Wout[i];
    if (tid < 2) lbo[tid] = bout[tid];
    __syncthreads();

    int ct = wv;                        // hidden col tile 0..15
    int coln = lane & 15;
    int rofs = (lane < 16) ? 0 : 8;

    for (int l = 0; l < L_; ++l) {
        if (tid < 64) {
            int b = rbase + tid;
            const float* p1 = regc + (((size_t)b * 4 + sl[tid * 4 + 1]) * L_ + l) * 2;
            const float* p2 = regc + (((size_t)b * 4 + sl[tid * 4 + 2]) * L_ + l) * 2;
            cf[tid * 4 + 0] = p1[0]; cf[tid * 4 + 1] = p1[1];
            cf[tid * 4 + 2] = p2[0]; cf[tid * 4 + 3] = p2[1];
        }
        __syncthreads();

        float nh[4][8];
        for (int rt = 0; rt < 4; ++rt) {
            v8f gz = {}, gr = {}, gn = {};
            const bf16* ap = hbf + (size_t)rt * 16 * 256;
            for (int g = 0; g < 3; ++g) {
                v8f acc = {};
                const bf16* bp = WhT + (size_t)(g * 256 + ct * 16) * 256;
                if (g < 2)   // warm L2/WGP$ for next gate slab while WMMAs run
                    __builtin_prefetch(WhT + (size_t)((g + 1) * 256 + ct * 16) * 256, 0, 0);
                for (int k0 = 0; k0 < 256; k0 += 32) {
                    v16bf a = load_a16x32(ap + k0, 256, lane);
                    v16bf b = load_b32x16(bp + k0, 256, lane);
                    acc = wmma_bf16(a, b, acc);
                }
                if (g == 0) gz = acc; else if (g == 1) gr = acc; else gn = acc;
            }
            int colh = ct * 16 + coln;
#pragma unroll
            for (int i = 0; i < 8; ++i) {
                int r = rt * 16 + rofs + i;
                float c10 = cf[r * 4 + 0], c11 = cf[r * 4 + 1];
                float c20 = cf[r * 4 + 2], c21 = cf[r * 4 + 3];
                int jz = colh, jr = colh + 256, jn = colh + 512;
                float gxz = c10 * lA1[jz] + c11 * lA1[G3_ + jz] + c20 * lA2[jz] + c21 * lA2[G3_ + jz] + lbg[jz];
                float gxr = c10 * lA1[jr] + c11 * lA1[G3_ + jr] + c20 * lA2[jr] + c21 * lA2[G3_ + jr] + lbg[jr];
                float gxn = c10 * lA1[jn] + c11 * lA1[G3_ + jn] + c20 * lA2[jn] + c21 * lA2[G3_ + jn] + lbg[jn];
                float z  = sigm(gxz + gz[i]);
                float rr = sigm(gxr + gr[i]);
                float n  = tanhf(gxn + rr * gn[i]);
                float h  = hf[(size_t)r * 256 + colh];
                nh[rt][i] = (1.f - z) * h + z * n;
            }
        }
        __syncthreads();   // all reads of h done before writes
        for (int rt = 0; rt < 4; ++rt) {
            int colh = ct * 16 + coln;
#pragma unroll
            for (int i = 0; i < 8; ++i) {
                int r = rt * 16 + rofs + i;
                hf[(size_t)r * 256 + colh]  = nh[rt][i];
                hbf[(size_t)r * 256 + colh] = (bf16)nh[rt][i];
            }
        }
        __syncthreads();

        // add-bit logits -> res_add coefficients (p0, p1)
        if (tid < 128) {
            int r = tid >> 1, c = tid & 1;
            float s = lbo[c];
            for (int k = 0; k < 256; ++k) s += hf[(size_t)r * 256 + k] * lWo[k * 2 + c];
            lg[r * 2 + c] = s;
        }
        __syncthreads();
        if (tid < 64) {
            float l0 = lg[tid * 2], l1 = lg[tid * 2 + 1];
            float m = fmaxf(l0, l1);
            float e0 = __expf(l0 - m), e1 = __expf(l1 - m);
            float p1 = e1 / (e0 + e1);
            size_t o = ((size_t)(rbase + tid) * L_ + l) * 2;
            resadd[o] = 1.f - p1; resadd[o + 1] = p1;
        }
        __syncthreads();
    }

    // pass 1: result coefficients (op select over val_s1 / res_add / shift)
    for (int i = tid; i < 64 * L_; i += blockDim.x) {
        int r = i / L_, l = i % L_;
        int b = rbase + r;
        int op = sl[r * 4 + 0], s1 = sl[r * 4 + 1];
        size_t o = ((size_t)b * L_ + l) * 2;
        float c0, c1;
        if (op == 0) {
            const float* p = regc + (((size_t)b * 4 + s1) * L_ + l) * 2;
            c0 = p[0]; c1 = p[1];
        } else if (op == 1) {
            c0 = resadd[o]; c1 = resadd[o + 1];
        } else {
            if (l == 0) { c0 = 1.f; c1 = 0.f; }
            else {
                const float* p = regc + (((size_t)b * 4 + s1) * L_ + (l - 1)) * 2;
                c0 = p[0]; c1 = p[1];
            }
        }
        resadd[o] = c0; resadd[o + 1] = c1;
    }
    __threadfence_block();
    __syncthreads();
    // pass 2: write selected destination register
    for (int i = tid; i < 64 * L_; i += blockDim.x) {
        int r = i / L_, l = i % L_;
        int b = rbase + r;
        int d = sl[r * 4 + 3];
        size_t o = ((size_t)b * L_ + l) * 2;
        float* q = regc + (((size_t)b * 4 + d) * L_ + l) * 2;
        q[0] = resadd[o]; q[1] = resadd[o + 1];
    }
}

// final logits: regs[:,2] @ bit[:2]^T via 2x2 Gram matrix
__global__ void k_final(const float* __restrict__ regc, const float* __restrict__ Gm,
                        float* __restrict__ out) {
    int i = blockIdx.x * blockDim.x + threadIdx.x;
    if (i >= B_ * L_ * 2) return;
    int j = i & 1, l = (i >> 1) % L_, b = i / (2 * L_);
    const float* c = regc + (((size_t)b * 4 + 2) * L_ + l) * 2;
    out[i] = c[0] * Gm[0 * 2 + j] + c[1] * Gm[1 * 2 + j];
}

// ---------------------------------------------------------------------------
extern "C" void kernel_launch(void* const* d_in, const int* in_sizes, int n_in,
                              void* d_out, int out_size, void* d_ws, size_t ws_size,
                              hipStream_t stream) {
    (void)in_sizes; (void)n_in; (void)out_size; (void)ws_size;

    const int*   r0       = (const int*)d_in[0];
    const int*   r1       = (const int*)d_in[1];
    const float* bit      = (const float*)d_in[2];
    const float* step_emb = (const float*)d_in[3];
    struct Layer { const float *Wqkv, *bqkv, *Wo, *bo, *ln1g, *ln1b, *W1, *b1, *W2, *b2, *ln2g, *ln2b; } ly[2];
    for (int i = 0; i < 2; ++i) {
        const float** pp = (const float**)&ly[i];
        for (int j = 0; j < 12; ++j) pp[j] = (const float*)d_in[4 + i * 12 + j];
    }
    int ix = 28;
    const float* Wop  = (const float*)d_in[ix++];
    const float* bop  = (const float*)d_in[ix++];
    const float* Ws1  = (const float*)d_in[ix++];
    const float* bs1  = (const float*)d_in[ix++];
    const float* Ws2  = (const float*)d_in[ix++];
    const float* bs2  = (const float*)d_in[ix++];
    const float* Wd   = (const float*)d_in[ix++];
    const float* bd   = (const float*)d_in[ix++];
    const float* Wx   = (const float*)d_in[ix++];
    const float* Wh   = (const float*)d_in[ix++];
    const float* bgru = (const float*)d_in[ix++];
    const float* Wout = (const float*)d_in[ix++];
    const float* bout = (const float*)d_in[ix++];

    // ---- workspace carve ----
    char* w = (char*)d_ws; size_t off = 0;
    auto alloc = [&](size_t bytes) { void* p = w + off; off = (off + bytes + 255) & ~(size_t)255; return p; };
    bf16* wqkv_b[2], *wo_b[2], *w1_b[2], *w2_b[2];
    for (int i = 0; i < 2; ++i) {
        wqkv_b[i] = (bf16*)alloc((size_t)G3_ * D_ * 2);
        wo_b[i]   = (bf16*)alloc((size_t)D_ * D_ * 2);
        w1_b[i]   = (bf16*)alloc((size_t)D_ * D_ * 2);
        w2_b[i]   = (bf16*)alloc((size_t)D_ * D_ * 2);
    }
    bf16*  whT    = (bf16*)alloc((size_t)G3_ * D_ * 2);
    float* A1     = (float*)alloc((size_t)2 * G3_ * 4);
    float* A2     = (float*)alloc((size_t)2 * G3_ * 4);
    float* Gm     = (float*)alloc(4 * 4);
    float* regc   = (float*)alloc((size_t)B_ * 4 * L_ * 2 * 4);
    bf16*  xbf    = (bf16*)alloc((size_t)B_ * 4 * D_ * 2);
    float* xres   = (float*)alloc((size_t)B_ * 4 * D_ * 4);
    bf16*  qkvb   = (bf16*)alloc((size_t)B_ * 4 * G3_ * 2);
    bf16*  atto   = (bf16*)alloc((size_t)B_ * 4 * D_ * 2);
    float* gf     = (float*)alloc((size_t)B_ * 4 * D_ * 4);
    bf16*  ffb    = (bf16*)alloc((size_t)B_ * 4 * D_ * 2);
    float* lg15   = (float*)alloc((size_t)B_ * 16 * 4);
    int*   selbuf = (int*)alloc((size_t)B_ * 4 * 4);
    float* resadd = (float*)alloc((size_t)B_ * L_ * 2 * 4);

    float* out        = (float*)d_out;
    float* out_logits = out;
    float* out_ops    = out + (size_t)B_ * L_ * 2;
    float* out_dests  = out_ops + (size_t)TSTEPS * B_;

    // ---- weight prep ----
    for (int i = 0; i < 2; ++i) {
        k_cvt<<<(G3_ * D_ + 255) / 256, 256, 0, stream>>>(ly[i].Wqkv, wqkv_b[i], G3_ * D_);
        k_cvt<<<(D_ * D_ + 255) / 256, 256, 0, stream>>>(ly[i].Wo, wo_b[i], D_ * D_);
        k_cvt<<<(D_ * D_ + 255) / 256, 256, 0, stream>>>(ly[i].W1, w1_b[i], D_ * D_);
        k_cvt<<<(D_ * D_ + 255) / 256, 256, 0, stream>>>(ly[i].W2, w2_b[i], D_ * D_);
    }
    k_transpose_cvt<<<(G3_ * D_ + 255) / 256, 256, 0, stream>>>(Wh, whT, D_, G3_);
    k_prep_small<<<3, 256, 0, stream>>>(bit, Wx, A1, A2, Gm);
    k_init_regs<<<(B_ * L_ + 255) / 256, 256, 0, stream>>>(r0, r1, regc);

    auto gemm = [&](const bf16* X, const bf16* WT, const float* bias,
                    float* Yf, bf16* Yb, int M, int N, int K, int relu) {
        int waves = (M >> 4) * (N >> 4);
        int blocks = (waves * 32 + 255) / 256;
        k_gemm_wt<<<blocks, 256, 0, stream>>>(X, WT, bias, Yf, Yb, M, N, K, relu);
    };

    const int M = B_ * 4;
    for (int t = 0; t < TSTEPS; ++t) {
        k_state<<<(M * 32 + 255) / 256, 256, 0, stream>>>(regc, bit, step_emb + (size_t)t * D_, xres, xbf);
        for (int i = 0; i < 2; ++i) {
            gemm(xbf, wqkv_b[i], ly[i].bqkv, nullptr, qkvb, M, G3_, D_, 0);
            k_attn<<<(B_ * NH_ * 4 + 255) / 256, 256, 0, stream>>>(qkvb, atto);
            gemm(atto, wo_b[i], ly[i].bo, gf, nullptr, M, D_, D_, 0);
            k_res_ln<<<(M * 32 + 255) / 256, 256, 0, stream>>>(gf, ly[i].ln1g, ly[i].ln1b, xres, xbf);
            gemm(xbf, w1_b[i], ly[i].b1, nullptr, ffb, M, D_, D_, 1);
            gemm(ffb, w2_b[i], ly[i].b2, gf, nullptr, M, D_, D_, 0);
            k_res_ln<<<(M * 32 + 255) / 256, 256, 0, stream>>>(gf, ly[i].ln2g, ly[i].ln2b, xres, xbf);
        }
        k_decode<<<(B_ * 32 + 255) / 256, 256, 0, stream>>>(xres, Wop, bop, Ws1, bs1, Ws2, bs2, Wd, bd, lg15);
        k_select<<<(B_ + 255) / 256, 256, 0, stream>>>(lg15, selbuf, out_ops, out_dests, t);
        k_gru<<<16, 512, (size_t)GRU_LDS_BYTES, stream>>>(regc, selbuf, whT, A1, A2, bgru, Wout, bout, resadd);
    }
    k_final<<<(B_ * L_ * 2 + 255) / 256, 256, 0, stream>>>(regc, Gm, out_logits);
}
